// Q8LinearLora_64879775973398
// MI455X (gfx1250) — 8-bit matrix operand order repaired and checked by reading, compiled, not run on hardware
//
#include <hip/hip_runtime.h>
#include <math.h>

typedef __attribute__((ext_vector_type(8))) int v8i;

#define M_TOK 8192
#define IN_F  4096
#define OUT_F 4096
#define R_LORA 16

#define K1_KC 512

__device__ __forceinline__ float fma4b(float a, int w, float4 v) {
  a = fmaf((float)((signed char)(w)),       v.x, a);
  a = fmaf((float)((signed char)(w >> 8)),  v.y, a);
  a = fmaf((float)((signed char)(w >> 16)), v.z, a);
  a = fmaf((float)((signed char)(w >> 24)), v.w, a);
  return a;
}

__global__ __launch_bounds__(256) void lora_xa_kernel(
    const signed char* __restrict__ x, const float* __restrict__ x_scales,
    const float* __restrict__ lora_a, float* __restrict__ T)
{
  __shared__ __align__(16) float A1[16 * K1_KC];

  const int t      = threadIdx.x;
  const int wave   = t >> 5;
  const int lane   = t & 31;
  const int laneHi = lane >> 4;
  const int r      = lane & 15;
  const int mBase  = blockIdx.x * 64;

  float acc[4] = {0.f, 0.f, 0.f, 0.f};

  for (int kc = 0; kc < IN_F; kc += K1_KC) {
    __syncthreads();
    for (int i = t; i < 16 * K1_KC / 4; i += 256) {
      int rr = i >> 7;
      int kq = i & 127;
      ((float4*)A1)[i] = *(const float4*)(lora_a + rr * IN_F + kc + kq * 4);
    }
    __syncthreads();
    #pragma unroll
    for (int p = 0; p < 4; ++p) {
      const int row = mBase + wave * 8 + p * 2 + laneHi;
      const signed char* xrow = x + (size_t)row * IN_F + kc;
      const float* ar = A1 + r * K1_KC;
      float a = acc[p];
      for (int k = 0; k < K1_KC; k += 16) {
        int4  xw = *(const int4*)(xrow + k);
        float4 a0 = *(const float4*)(ar + k);
        float4 a1 = *(const float4*)(ar + k + 4);
        float4 a2 = *(const float4*)(ar + k + 8);
        float4 a3 = *(const float4*)(ar + k + 12);
        a = fma4b(a, xw.x, a0);
        a = fma4b(a, xw.y, a1);
        a = fma4b(a, xw.z, a2);
        a = fma4b(a, xw.w, a3);
      }
      acc[p] = a;
    }
  }

  #pragma unroll
  for (int p = 0; p < 4; ++p) {
    const int row = mBase + wave * 8 + p * 2 + laneHi;
    T[row * R_LORA + r] = x_scales[row] * acc[p];
  }
}

#define BM  128
#define BN  128
#define BK  64
#define LDA 80
#define ABYTES (BM * LDA)

__device__ __forceinline__ unsigned lds_addr(const void* p) {
  return (unsigned)(uintptr_t)p;
}

__device__ __forceinline__ void async_copy_b128(unsigned lds_off,
                                                const signed char* gaddr) {
  asm volatile("global_load_async_to_lds_b128 %0, %1, off"
               :: "v"(lds_off), "v"(gaddr) : "memory");
}

__device__ __forceinline__ float dot4f(float4 a, float4 b) {
  return fmaf(a.x, b.x, fmaf(a.y, b.y, fmaf(a.z, b.z, a.w * b.w)));
}

__device__ __forceinline__ float gelu_tanh(float v) {
  float u = 0.7978845608028654f * fmaf(0.044715f * v, v * v, v);
  return 0.5f * v * (1.0f + tanhf(u));
}

__global__ __launch_bounds__(256) void q8_wmma_kernel(
    const signed char* __restrict__ x, const signed char* __restrict__ w,
    const float* __restrict__ bias, const float* __restrict__ x_scales,
    const float* __restrict__ scales, const float* __restrict__ lora_b,
    const float* __restrict__ T, const int* __restrict__ fuse_p,
    float* __restrict__ out)
{
  __shared__ __align__(16) signed char As[2 * ABYTES];
  __shared__ __align__(16) signed char Bs[2 * ABYTES];
  __shared__ __align__(16) float Ts [BM * R_LORA];
  __shared__ __align__(16) float Bls[BN * R_LORA];
  __shared__ float Xss[BM];

  const int t      = threadIdx.x;
  const int lane   = t & 31;
  const int wave   = t >> 5;
  const int waveM  = wave & 3;
  const int waveN  = wave >> 2;
  const int laneHi = lane >> 4;
  const int lane16 = lane & 15;
  const int mBlock = blockIdx.y * BM;
  const int oBlock = blockIdx.x * BN;

  for (int i = t; i < BM * R_LORA / 4; i += 256)
    ((float4*)Ts)[i] = ((const float4*)(T + (size_t)mBlock * R_LORA))[i];
  for (int i = t; i < BN * R_LORA / 4; i += 256)
    ((float4*)Bls)[i] = ((const float4*)(lora_b + (size_t)oBlock * R_LORA))[i];
  if (t < BM) Xss[t] = x_scales[mBlock + t];

  const int row0 = t >> 2;
  const int c0   = t & 3;
  const int row1 = row0 + 64;
  const signed char* xg0 = x + (size_t)(mBlock + row0) * IN_F + c0 * 16;
  const signed char* xg1 = x + (size_t)(mBlock + row1) * IN_F + c0 * 16;
  const signed char* wg0 = w + (size_t)(oBlock + row0) * IN_F + c0 * 16;
  const signed char* wg1 = w + (size_t)(oBlock + row1) * IN_F + c0 * 16;
  const unsigned aOff0 = lds_addr(As) + row0 * LDA + c0 * 16;
  const unsigned aOff1 = lds_addr(As) + row1 * LDA + c0 * 16;
  const unsigned bOff0 = lds_addr(Bs) + row0 * LDA + c0 * 16;
  const unsigned bOff1 = lds_addr(Bs) + row1 * LDA + c0 * 16;

  v8i acc[2][4];
  #pragma unroll
  for (int fm = 0; fm < 2; ++fm)
    #pragma unroll
    for (int fn = 0; fn < 4; ++fn)
      acc[fm][fn] = (v8i){0, 0, 0, 0, 0, 0, 0, 0};

  const int NK = IN_F / BK;

  {
    async_copy_b128(aOff0, xg0);
    async_copy_b128(aOff1, xg1);
    async_copy_b128(bOff0, wg0);
    async_copy_b128(bOff1, wg1);
  }

  for (int kt = 0; kt < NK; ++kt) {
    const int buf = kt & 1;
    __syncthreads();
    if (kt + 1 < NK) {
      const unsigned sel = (unsigned)((buf ^ 1) * ABYTES);
      const int koff = (kt + 1) * BK;
      async_copy_b128(aOff0 + sel, xg0 + koff);
      async_copy_b128(aOff1 + sel, xg1 + koff);
      async_copy_b128(bOff0 + sel, wg0 + koff);
      async_copy_b128(bOff1 + sel, wg1 + koff);
      asm volatile("s_wait_asynccnt 0x4" ::: "memory");
    } else {
      asm volatile("s_wait_asynccnt 0x0" ::: "memory");
    }
    __syncthreads();

    const signed char* Ab = As + buf * ABYTES;
    const signed char* Bb = Bs + buf * ABYTES;

    v8i af[2];
    #pragma unroll
    for (int fm = 0; fm < 2; ++fm) {
      const signed char* ap = Ab + (waveM * 32 + fm * 16 + lane16) * LDA + laneHi * 16;
      int2 q0 = *(const int2*)(ap);
      int2 q1 = *(const int2*)(ap + 8);
      int2 q2 = *(const int2*)(ap + 32);
      int2 q3 = *(const int2*)(ap + 40);
      v8i a;
      a[0] = q0.x; a[1] = q0.y; a[2] = q1.x; a[3] = q1.y;
      a[4] = q2.x; a[5] = q2.y; a[6] = q3.x; a[7] = q3.y;
      af[fm] = a;
    }
    v8i bf[4];
    #pragma unroll
    for (int fn = 0; fn < 4; ++fn) {
      const signed char* bp = Bb + (waveN * 64 + fn * 16 + lane16) * LDA + laneHi * 16;
      int4 p0 = *(const int4*)(bp);
      int4 p1 = *(const int4*)(bp + 32);
      v8i b;
      b[0] = p0.x; b[1] = p0.y; b[2] = p0.z; b[3] = p0.w;
      b[4] = p1.x; b[5] = p1.y; b[6] = p1.z; b[7] = p1.w;
      bf[fn] = b;
    }

    #pragma unroll
    for (int fm = 0; fm < 2; ++fm)
      #pragma unroll
      for (int fn = 0; fn < 4; ++fn)
        acc[fm][fn] = __builtin_amdgcn_wmma_i32_16x16x64_iu8(
             true, af[fm],  true, bf[fn],
            acc[fm][fn],  false,  false);
  }

  const int fuse = *fuse_p;
  float4 bl[4][4];
  float  so[4], bo[4];
  int    nL[4];
  #pragma unroll
  for (int fn = 0; fn < 4; ++fn) {
    nL[fn] = waveN * 64 + fn * 16 + lane16;
    const int o = oBlock + nL[fn];
    so[fn] = scales[o];
    bo[fn] = bias[o];
    const float* bp = Bls + nL[fn] * R_LORA;
    bl[fn][0] = *(const float4*)(bp);
    bl[fn][1] = *(const float4*)(bp + 4);
    bl[fn][2] = *(const float4*)(bp + 8);
    bl[fn][3] = *(const float4*)(bp + 12);
  }

  #pragma unroll
  for (int fm = 0; fm < 2; ++fm) {
    #pragma unroll
    for (int j = 0; j < 8; ++j) {
      const int mL = waveM * 32 + fm * 16 + laneHi * 8 + j;
      const size_t m = (size_t)(mBlock + mL);
      const float xs = Xss[mL];
      const float* tp = Ts + mL * R_LORA;
      float4 t0 = *(const float4*)(tp);
      float4 t1 = *(const float4*)(tp + 4);
      float4 t2 = *(const float4*)(tp + 8);
      float4 t3 = *(const float4*)(tp + 12);
      #pragma unroll
      for (int fn = 0; fn < 4; ++fn) {
        float l = dot4f(t0, bl[fn][0]) + dot4f(t1, bl[fn][1]) +
                  dot4f(t2, bl[fn][2]) + dot4f(t3, bl[fn][3]);
        float v = (float)acc[fm][fn][j] * xs * so[fn] + l + bo[fn];
        if (fuse) v = gelu_tanh(v);
        out[m * OUT_F + oBlock + nL[fn]] = v;
      }
    }
  }
}

extern "C" void kernel_launch(void* const* d_in, const int* in_sizes, int n_in,
                              void* d_out, int out_size, void* d_ws, size_t ws_size,
                              hipStream_t stream) {
  const signed char* x        = (const signed char*)d_in[0];
  const signed char* weight   = (const signed char*)d_in[1];
  const float*       bias     = (const float*)d_in[2];
  const float*       x_scales = (const float*)d_in[3];
  const float*       scales   = (const float*)d_in[4];
  const float*       lora_a   = (const float*)d_in[5];
  const float*       lora_b   = (const float*)d_in[6];
  const int*         fuse     = (const int*)d_in[7];
  float*             out      = (float*)d_out;
  float*             T        = (float*)d_ws;

  lora_xa_kernel<<<dim3(M_TOK / 64), dim3(256), 0, stream>>>(x, x_scales, lora_a, T);

  dim3 grid2(OUT_F / BN, M_TOK / BM);
  q8_wmma_kernel<<<grid2, dim3(256), 0, stream>>>(
      x, weight, bias, x_scales, scales, lora_b, T, fuse, out);

  (void)in_sizes; (void)n_in; (void)out_size; (void)ws_size;
}
